// SwinLatentTransformer_43422119362773
// MI455X (gfx1250) — compile-verified, matrix operand order checked by reading, not run on hardware
//
#include <hip/hip_runtime.h>
#include <hip/hip_bf16.h>

typedef __attribute__((ext_vector_type(16))) _Float16 v16h;
typedef __attribute__((ext_vector_type(8)))  float    v8f;

// ---------------------------------------------------------------------------
// Constants for this problem instance
// ---------------------------------------------------------------------------
#define BB    8
#define HH    128
#define WWIM  128
#define CC    256
#define NHD   8
#define HD    32
#define WS    8
#define SS    4
#define NTOK  64            // tokens per window (8x8)
#define NWIN  2048          // B * (128/8)^2
#define MLPD  1024
#define QSCALE 0.17677669529663687f  // 1/sqrt(32)

// ---------------------------------------------------------------------------
// WMMA helpers (CDNA5 16x16x32 f16 -> f32)
// ---------------------------------------------------------------------------
__device__ __forceinline__ v8f wmma_f16(v16h a, v16h b, v8f c) {
  // 8 args: (neg_a, A, neg_b, B, c_mod, C, reuse_a, reuse_b)
  return __builtin_amdgcn_wmma_f32_16x16x32_f16(false, a, false, b, (short)0, c,
                                                false, false);
}

// Load a 16x32 f16 A/B fragment from an LDS tile (row-major, `ld` halfs/row).
// ISA layout: lane = {idx(0..15), half(0..1)}, elems 0..7 = row[k0+8h .. +7],
// elems 8..15 = row[k0+16+8h .. +7].
__device__ __forceinline__ v16h ld_frag(const _Float16* tile, int ld, int r0, int k0) {
  int lane = threadIdx.x & 31;
  const _Float16* p = tile + (r0 + (lane & 15)) * ld + k0 + ((lane >> 4) << 3);
  v16h f;
#pragma unroll
  for (int i = 0; i < 8; ++i) { f[i] = p[i]; f[i + 8] = p[16 + i]; }
  return f;
}

// Async global -> LDS tile staging (16B per lane per issue, ASYNCcnt path).
// GV mode: 64-bit global address in VGPR pair, LDS byte offset in one VGPR
// (low 32 bits of the generic pointer to __shared__).
__device__ __forceinline__ void copy_g2l_async(_Float16* dst, int dstLd,
                                               const _Float16* src, int srcLd,
                                               int rows, int cols8) {
  int total = rows * cols8;
  for (int i = (int)threadIdx.x; i < total; i += (int)blockDim.x) {
    int rr = i / cols8, cc = i - rr * cols8;
    unsigned lds = (unsigned)(unsigned long long)(dst + rr * dstLd + cc * 8);
    const _Float16* g = src + (size_t)rr * srcLd + cc * 8;
    asm volatile("global_load_async_to_lds_b128 %0, %1, off"
                 :: "v"(lds), "v"(g)
                 : "memory");
  }
}

__device__ __forceinline__ void async_wait0() {
  asm volatile("s_wait_asynccnt 0" ::: "memory");
}

__device__ __forceinline__ int region_lab(int v) {
  return v < (HH - WS) ? 0 : (v < (HH - SS) ? 1 : 2);
}

// ---------------------------------------------------------------------------
// K0: transpose + convert all weights to f16 (B^T layout for WMMA B-frags)
// ---------------------------------------------------------------------------
__global__ void prep_weights(const float* __restrict__ wqkv,
                             const float* __restrict__ wproj,
                             const float* __restrict__ wmlp1,
                             const float* __restrict__ wmlp2,
                             _Float16* __restrict__ wqkvT,
                             _Float16* __restrict__ wprojT,
                             _Float16* __restrict__ wmlp1T,
                             _Float16* __restrict__ wmlp2T) {
  int i = blockIdx.x * blockDim.x + threadIdx.x;
  if (i < 768 * 256)  wqkvT[i]  = (_Float16)wqkv [(i & 255) * 768  + (i >> 8)];
  if (i < 256 * 256)  wprojT[i] = (_Float16)wproj[(i & 255) * 256  + (i >> 8)];
  if (i < 1024 * 256) wmlp1T[i] = (_Float16)wmlp1[(i & 255) * 1024 + (i >> 8)];
  if (i < 256 * 1024) wmlp2T[i] = (_Float16)wmlp2[(i & 1023) * 256 + (i >> 10)];
}

// ---------------------------------------------------------------------------
// K1: LayerNorm1 fused with cyclic shift (-4,-4) + window partition -> f16
// ---------------------------------------------------------------------------
__global__ void __launch_bounds__(64) ln1_window(const float* __restrict__ xin,
                                                 const float* __restrict__ g,
                                                 const float* __restrict__ bt,
                                                 _Float16* __restrict__ xw) {
  int win = blockIdx.x, t = threadIdx.x;
  int b = win >> 8, wh = (win >> 4) & 15, ww = win & 15;
  int yi = t >> 3, xi = t & 7;
  int y = (wh * 8 + yi + SS) & (HH - 1);
  int x = (ww * 8 + xi + SS) & (WWIM - 1);
  const float4* src =
      (const float4*)(xin + ((((size_t)b << 14) + (y << 7) + x) << 8));
  float s = 0.f, s2 = 0.f;
  for (int i = 0; i < 64; ++i) {
    float4 v = src[i];
    s  += v.x + v.y + v.z + v.w;
    s2 += v.x * v.x + v.y * v.y + v.z * v.z + v.w * v.w;
  }
  float mean = s * (1.f / 256.f);
  float var  = s2 * (1.f / 256.f) - mean * mean;
  float rs   = rsqrtf(var + 1e-5f);
  _Float16* dst = xw + ((size_t)win * NTOK + t) * CC;
  for (int i = 0; i < 64; ++i) {
    float4 v = src[i];
    dst[4 * i + 0] = (_Float16)((v.x - mean) * rs * g[4 * i + 0] + bt[4 * i + 0]);
    dst[4 * i + 1] = (_Float16)((v.y - mean) * rs * g[4 * i + 1] + bt[4 * i + 1]);
    dst[4 * i + 2] = (_Float16)((v.z - mean) * rs * g[4 * i + 2] + bt[4 * i + 2]);
    dst[4 * i + 3] = (_Float16)((v.w - mean) * rs * g[4 * i + 3] + bt[4 * i + 3]);
  }
}

// ---------------------------------------------------------------------------
// K2: QKV GEMM. block = (window, 64-col tile of 768); 8 waves, 64x64 out tile.
// ---------------------------------------------------------------------------
__global__ void __launch_bounds__(256) qkv_gemm(const _Float16* __restrict__ xw,
                                                const _Float16* __restrict__ wqkvT,
                                                const float* __restrict__ bqkv,
                                                _Float16* __restrict__ qb,
                                                _Float16* __restrict__ kb,
                                                _Float16* __restrict__ vb) {
  __shared__ __align__(16) _Float16 As[64 * 264];
  __shared__ __align__(16) _Float16 Bs[64 * 264];
  int blk = blockIdx.x;
  int win = blk / 12, ct = blk - win * 12;
  copy_g2l_async(As, 264, xw + (size_t)win * NTOK * CC, 256, 64, 32);
  copy_g2l_async(Bs, 264, wqkvT + (size_t)ct * 64 * 256, 256, 64, 32);
  async_wait0();
  __syncthreads();
  int wv = threadIdx.x >> 5, lane = threadIdx.x & 31;
  int rsub = wv >> 1, csub0 = (wv & 1) * 2;
  v8f acc0 = {}, acc1 = {};
  for (int kk = 0; kk < 8; ++kk) {
    v16h a  = ld_frag(As, 264, rsub * 16, kk * 32);
    v16h b0 = ld_frag(Bs, 264, csub0 * 16, kk * 32);
    v16h b1 = ld_frag(Bs, 264, (csub0 + 1) * 16, kk * 32);
    acc0 = wmma_f16(a, b0, acc0);
    acc1 = wmma_f16(a, b1, acc1);
  }
  int n = lane & 15, hf = lane >> 4;
#pragma unroll
  for (int s = 0; s < 2; ++s) {
    v8f acc = s ? acc1 : acc0;
    int col = ct * 64 + (csub0 + s) * 16 + n;
    float bias = bqkv[col];
#pragma unroll
    for (int r = 0; r < 8; ++r) {
      int m = rsub * 16 + r + hf * 8;
      float val = acc[r] + bias;
      if (col < 256) {
        int h = col >> 5, d = col & 31;
        qb[(((size_t)win * NHD + h) * NTOK + m) * HD + d] =
            (_Float16)(val * QSCALE);
      } else if (col < 512) {
        int cc = col - 256, h = cc >> 5, d = cc & 31;
        kb[(((size_t)win * NHD + h) * NTOK + m) * HD + d] = (_Float16)val;
      } else {
        int cc = col - 512, h = cc >> 5, d = cc & 31;
        vb[(((size_t)win * NHD + h) * HD + d) * NTOK + m] = (_Float16)val;
      }
    }
  }
}

// ---------------------------------------------------------------------------
// K3: attention per (window, head). S = q k^T, + bias + mask, softmax, O = P V
// ---------------------------------------------------------------------------
__global__ void __launch_bounds__(128) attention(const _Float16* __restrict__ qb,
                                                 const _Float16* __restrict__ kb,
                                                 const _Float16* __restrict__ vb,
                                                 const float* __restrict__ btab,
                                                 _Float16* __restrict__ ao) {
  __shared__ __align__(16) _Float16 Qs[64 * 40];
  __shared__ __align__(16) _Float16 Ks[64 * 40];
  __shared__ __align__(16) _Float16 Vt[32 * 72];
  __shared__ float Sf[64 * 65];
  __shared__ __align__(16) _Float16 Ps[64 * 72];
  int wh_idx = blockIdx.x;
  int win = wh_idx >> 3, h = wh_idx & 7;
  copy_g2l_async(Qs, 40, qb + (size_t)wh_idx * 2048, 32, 64, 4);
  copy_g2l_async(Ks, 40, kb + (size_t)wh_idx * 2048, 32, 64, 4);
  copy_g2l_async(Vt, 72, vb + (size_t)wh_idx * 2048, 64, 32, 8);
  async_wait0();
  __syncthreads();
  int wv = threadIdx.x >> 5, lane = threadIdx.x & 31;
  int n = lane & 15, hf = lane >> 4;
  int wimg = win & 255;
  int wy0 = ((wimg >> 4) << 3), wx0 = ((wimg & 15) << 3);

  v16h aq = ld_frag(Qs, 40, wv * 16, 0);
#pragma unroll
  for (int c = 0; c < 4; ++c) {
    v16h bk = ld_frag(Ks, 40, c * 16, 0);
    v8f z = {};
    v8f sc = wmma_f16(aq, bk, z);
    int j = c * 16 + n;
    int yj = j >> 3, xj = j & 7;
    int lj = region_lab(wy0 + yj) * 3 + region_lab(wx0 + xj);
#pragma unroll
    for (int r = 0; r < 8; ++r) {
      int i = wv * 16 + r + hf * 8;
      int yi = i >> 3, xi = i & 7;
      int rel = (yi - yj + 7) * 15 + (xi - xj + 7);
      float bv = btab[rel * NHD + h];
      int li = region_lab(wy0 + yi) * 3 + region_lab(wx0 + xi);
      float mv = (li == lj) ? 0.f : -100.f;
      Sf[i * 65 + j] = sc[r] + bv + mv;
    }
  }
  __syncthreads();
  if (threadIdx.x < 64) {
    int i = threadIdx.x;
    float mx = -1e30f;
    for (int j = 0; j < 64; ++j) mx = fmaxf(mx, Sf[i * 65 + j]);
    float sum = 0.f;
    for (int j = 0; j < 64; ++j) {
      float e = __expf(Sf[i * 65 + j] - mx);
      Sf[i * 65 + j] = e;
      sum += e;
    }
    float inv = 1.f / sum;
    for (int j = 0; j < 64; ++j) Ps[i * 72 + j] = (_Float16)(Sf[i * 65 + j] * inv);
  }
  __syncthreads();
#pragma unroll
  for (int c = 0; c < 2; ++c) {
    v8f acc = {};
#pragma unroll
    for (int kk = 0; kk < 2; ++kk) {
      v16h a = ld_frag(Ps, 72, wv * 16, kk * 32);
      v16h b = ld_frag(Vt, 72, c * 16, kk * 32);
      acc = wmma_f16(a, b, acc);
    }
#pragma unroll
    for (int r = 0; r < 8; ++r) {
      int m = wv * 16 + r + hf * 8;
      int d = c * 16 + n;
      ao[((size_t)win * NTOK + m) * CC + h * HD + d] = (_Float16)acc[r];
    }
  }
}

// ---------------------------------------------------------------------------
// K4: output projection + window reverse + un-shift + residual -> y in d_out
// ---------------------------------------------------------------------------
__global__ void __launch_bounds__(256) proj_res(const _Float16* __restrict__ ao,
                                                const _Float16* __restrict__ wprojT,
                                                const float* __restrict__ bproj,
                                                const float* __restrict__ xin,
                                                float* __restrict__ out) {
  __shared__ __align__(16) _Float16 As[64 * 264];
  __shared__ __align__(16) _Float16 Bs[64 * 264];
  int blk = blockIdx.x;
  int win = blk >> 2, ct = blk & 3;
  copy_g2l_async(As, 264, ao + (size_t)win * NTOK * CC, 256, 64, 32);
  copy_g2l_async(Bs, 264, wprojT + (size_t)ct * 64 * 256, 256, 64, 32);
  async_wait0();
  __syncthreads();
  int wv = threadIdx.x >> 5, lane = threadIdx.x & 31;
  int rsub = wv >> 1, csub0 = (wv & 1) * 2;
  v8f acc0 = {}, acc1 = {};
  for (int kk = 0; kk < 8; ++kk) {
    v16h a  = ld_frag(As, 264, rsub * 16, kk * 32);
    v16h b0 = ld_frag(Bs, 264, csub0 * 16, kk * 32);
    v16h b1 = ld_frag(Bs, 264, (csub0 + 1) * 16, kk * 32);
    acc0 = wmma_f16(a, b0, acc0);
    acc1 = wmma_f16(a, b1, acc1);
  }
  int n = lane & 15, hf = lane >> 4;
  int b = win >> 8, wh = (win >> 4) & 15, ww = win & 15;
#pragma unroll
  for (int s = 0; s < 2; ++s) {
    v8f acc = s ? acc1 : acc0;
    int col = ct * 64 + (csub0 + s) * 16 + n;
    float bias = bproj[col];
#pragma unroll
    for (int r = 0; r < 8; ++r) {
      int m = rsub * 16 + r + hf * 8;
      int yi = m >> 3, xi = m & 7;
      int y = (wh * 8 + yi + SS) & (HH - 1);
      int x = (ww * 8 + xi + SS) & (WWIM - 1);
      size_t pos = ((size_t)b << 14) + (y << 7) + x;
      out[pos * CC + col] = xin[pos * CC + col] + acc[r] + bias;
    }
  }
}

// ---------------------------------------------------------------------------
// K5: LayerNorm2 on y (flat token order) -> f16
// ---------------------------------------------------------------------------
__global__ void __launch_bounds__(64) ln2_kernel(const float* __restrict__ yin,
                                                 const float* __restrict__ g,
                                                 const float* __restrict__ bt,
                                                 _Float16* __restrict__ h2) {
  size_t tok = (size_t)blockIdx.x * 64 + threadIdx.x;
  const float4* src = (const float4*)(yin + tok * CC);
  float s = 0.f, s2 = 0.f;
  for (int i = 0; i < 64; ++i) {
    float4 v = src[i];
    s  += v.x + v.y + v.z + v.w;
    s2 += v.x * v.x + v.y * v.y + v.z * v.z + v.w * v.w;
  }
  float mean = s * (1.f / 256.f);
  float var  = s2 * (1.f / 256.f) - mean * mean;
  float rs   = rsqrtf(var + 1e-5f);
  _Float16* dst = h2 + tok * CC;
  for (int i = 0; i < 64; ++i) {
    float4 v = src[i];
    dst[4 * i + 0] = (_Float16)((v.x - mean) * rs * g[4 * i + 0] + bt[4 * i + 0]);
    dst[4 * i + 1] = (_Float16)((v.y - mean) * rs * g[4 * i + 1] + bt[4 * i + 1]);
    dst[4 * i + 2] = (_Float16)((v.z - mean) * rs * g[4 * i + 2] + bt[4 * i + 2]);
    dst[4 * i + 3] = (_Float16)((v.w - mean) * rs * g[4 * i + 3] + bt[4 * i + 3]);
  }
}

// ---------------------------------------------------------------------------
// K6: fused MLP: out += gelu(h2 @ W1 + b1) @ W2 + b2   (residual in d_out)
// ---------------------------------------------------------------------------
__global__ void __launch_bounds__(256) mlp_fused(const _Float16* __restrict__ h2,
                                                 const _Float16* __restrict__ w1T,
                                                 const float* __restrict__ b1,
                                                 const _Float16* __restrict__ w2T,
                                                 const float* __restrict__ b2,
                                                 float* __restrict__ out) {
  __shared__ __align__(16) _Float16 As[64 * 264];
  __shared__ __align__(16) _Float16 B1s[64 * 264];
  __shared__ __align__(16) _Float16 B2s[256 * 72];
  __shared__ __align__(16) _Float16 Ps[64 * 72];
  int rt = blockIdx.x;
  copy_g2l_async(As, 264, h2 + (size_t)rt * 64 * CC, 256, 64, 32);
  int wv = threadIdx.x >> 5, lane = threadIdx.x & 31;
  int n = lane & 15, hf = lane >> 4;
  int orow = wv & 3, ocg = wv >> 2;       // O-tile: row subtile, col group
  int r1 = wv >> 1, c10 = (wv & 1) * 2;   // H-tile assignment for GEMM1
  v8f zero = {};
  v8f oacc[8];
#pragma unroll
  for (int i = 0; i < 8; ++i) oacc[i] = zero;

  for (int hc = 0; hc < 16; ++hc) {
    __syncthreads();
    copy_g2l_async(B1s, 264, w1T + (size_t)hc * 64 * 256, 256, 64, 32);
    for (int i = (int)threadIdx.x; i < 256 * 8; i += 256) {
      int rr = i >> 3, cc = i & 7;
      unsigned lds = (unsigned)(unsigned long long)(B2s + rr * 72 + cc * 8);
      const _Float16* gsrc = w2T + (size_t)rr * 1024 + hc * 64 + cc * 8;
      asm volatile("global_load_async_to_lds_b128 %0, %1, off"
                   :: "v"(lds), "v"(gsrc)
                   : "memory");
    }
    // pull next chunk's W1 tile toward the caches while this one computes
    if (hc + 1 < 16)
      __builtin_prefetch(w1T + (size_t)(hc + 1) * 64 * 256 + threadIdx.x * 64, 0, 1);
    async_wait0();
    __syncthreads();
    // GEMM1: H chunk [64 x 64], K = 256
    v8f ha0 = zero, ha1 = zero;
    for (int kk = 0; kk < 8; ++kk) {
      v16h a  = ld_frag(As, 264, r1 * 16, kk * 32);
      v16h b0 = ld_frag(B1s, 264, c10 * 16, kk * 32);
      v16h bx = ld_frag(B1s, 264, (c10 + 1) * 16, kk * 32);
      ha0 = wmma_f16(a, b0, ha0);
      ha1 = wmma_f16(a, bx, ha1);
    }
#pragma unroll
    for (int s = 0; s < 2; ++s) {
      v8f ha = s ? ha1 : ha0;
      int hcol = (c10 + s) * 16 + n;
      float bias = b1[hc * 64 + hcol];
#pragma unroll
      for (int r = 0; r < 8; ++r) {
        int m = r1 * 16 + r + hf * 8;
        float hv = ha[r] + bias;
        float ge = 0.5f * hv * (1.0f + erff(hv * 0.70710678f));  // exact GELU
        Ps[m * 72 + hcol] = (_Float16)ge;
      }
    }
    __syncthreads();
    // GEMM2 accumulate: O[16 x 128] strip per wave, K-chunk = 64
#pragma unroll
    for (int kk = 0; kk < 2; ++kk) {
      v16h a = ld_frag(Ps, 72, orow * 16, kk * 32);
#pragma unroll
      for (int c = 0; c < 8; ++c) {
        v16h b = ld_frag(B2s, 72, (ocg * 8 + c) * 16, kk * 32);
        oacc[c] = wmma_f16(a, b, oacc[c]);
      }
    }
  }
  // epilogue: += residual (y already in out) + b2
#pragma unroll
  for (int c = 0; c < 8; ++c) {
    int col = (ocg * 8 + c) * 16 + n;
    float bias = b2[col];
#pragma unroll
    for (int r = 0; r < 8; ++r) {
      int m = orow * 16 + r + hf * 8;
      size_t tok = (size_t)rt * 64 + m;
      out[tok * CC + col] = out[tok * CC + col] + oacc[c][r] + bias;
    }
  }
}

// ---------------------------------------------------------------------------
// Launcher
// ---------------------------------------------------------------------------
extern "C" void kernel_launch(void* const* d_in, const int* in_sizes, int n_in,
                              void* d_out, int out_size, void* d_ws, size_t ws_size,
                              hipStream_t stream) {
  (void)in_sizes; (void)n_in; (void)out_size; (void)ws_size;
  const float* x     = (const float*)d_in[0];
  const float* n1g   = (const float*)d_in[3];
  const float* n1b   = (const float*)d_in[4];
  const float* wqkv  = (const float*)d_in[5];
  const float* bqkv  = (const float*)d_in[6];
  const float* btab  = (const float*)d_in[7];
  const float* wproj = (const float*)d_in[8];
  const float* bproj = (const float*)d_in[9];
  const float* n2g   = (const float*)d_in[10];
  const float* n2b   = (const float*)d_in[11];
  const float* w1    = (const float*)d_in[12];
  const float* b1    = (const float*)d_in[13];
  const float* w2    = (const float*)d_in[14];
  const float* b2    = (const float*)d_in[15];
  float* out = (float*)d_out;

  char* ws = (char*)d_ws;
  _Float16* wqkvT  = (_Float16*)(ws);
  _Float16* wprojT = (_Float16*)(ws + 393216);
  _Float16* wmlp1T = (_Float16*)(ws + 393216 + 131072);
  _Float16* wmlp2T = (_Float16*)(ws + 393216 + 131072 + 524288);
  const size_t base = 1572864;
  const size_t BUF  = 67108864;  // 2048*64*256 f16
  _Float16* bufA = (_Float16*)(ws + base);             // xw, later attn_out
  _Float16* bufQ = (_Float16*)(ws + base + BUF);       // q, later h2
  _Float16* bufK = (_Float16*)(ws + base + 2 * BUF);   // k
  _Float16* bufV = (_Float16*)(ws + base + 3 * BUF);   // v^T

  prep_weights<<<1024, 256, 0, stream>>>(wqkv, wproj, w1, w2,
                                         wqkvT, wprojT, wmlp1T, wmlp2T);
  ln1_window<<<NWIN, 64, 0, stream>>>(x, n1g, n1b, bufA);
  qkv_gemm<<<NWIN * 12, 256, 0, stream>>>(bufA, wqkvT, bqkv, bufQ, bufK, bufV);
  attention<<<NWIN * NHD, 128, 0, stream>>>(bufQ, bufK, bufV, btab, bufA);
  proj_res<<<NWIN * 4, 256, 0, stream>>>(bufA, wprojT, bproj, x, out);
  ln2_kernel<<<2048, 64, 0, stream>>>(out, n2g, n2b, bufQ);
  mlp_fused<<<2048, 256, 0, stream>>>(bufQ, wmlp1T, b1, wmlp2T, b2, out);
}
